// SparseMoeBlock_64510408785934
// MI455X (gfx1250) — matrix operand order repaired and checked by reading, not compiled, not run on hardware
//
#include <hip/hip_runtime.h>
#include <hip/hip_bf16.h>

// Problem constants (from reference): B=4, S=4096, D=1024, F=4096, E=8, K=2
#define T_TOK 16384
#define D_DIM 1024
#define F_DIM 4096
#define E_EXP 8
#define BM    64      // tokens per workgroup tile

typedef __attribute__((ext_vector_type(16))) __bf16 v16bf;
typedef __attribute__((ext_vector_type(8)))  float  v8f;

union FragB16 { v16bf v; uint4 q[2]; };

__device__ __forceinline__ unsigned short f32_to_bf16(float f) {
  unsigned u = __float_as_uint(f);
  unsigned r = (u + 0x7FFFu + ((u >> 16) & 1u)) >> 16;  // round-to-nearest-even
  return (unsigned short)r;
}

// ---------------- zero output ----------------
__global__ void zero_kernel(float4* __restrict__ p, long long n4) {
  long long i = (long long)blockIdx.x * blockDim.x + threadIdx.x;
  long long stride = (long long)gridDim.x * blockDim.x;
  float4 z = {0.f, 0.f, 0.f, 0.f};
  for (; i < n4; i += stride) p[i] = z;
}

// ---------------- fp32 -> bf16 (pairs, packed dword stores) ----------------
__global__ void cvt_bf16_kernel(const float2* __restrict__ s, unsigned* __restrict__ d,
                                long long n2) {
  long long i = (long long)blockIdx.x * blockDim.x + threadIdx.x;
  long long stride = (long long)gridDim.x * blockDim.x;
  for (; i < n2; i += stride) {
    float2 f = s[i];
    d[i] = (unsigned)f32_to_bf16(f.x) | ((unsigned)f32_to_bf16(f.y) << 16);
  }
}

// ---------------- router: logits, softmax, top-2 ----------------
__global__ void router_kernel(const float* __restrict__ x,
                              const float* __restrict__ gw,
                              float* __restrict__ probs_out,
                              int* __restrict__ tki, float* __restrict__ tkw) {
  const int lane = threadIdx.x & 31;
  const int t = blockIdx.x * 8 + (threadIdx.x >> 5);   // 8 waves per block, 1 token per wave
  const float* xr = x + (size_t)t * D_DIM;
  float acc[E_EXP];
#pragma unroll
  for (int e = 0; e < E_EXP; ++e) acc[e] = 0.f;
  for (int d = lane; d < D_DIM; d += 32) {
    float xv = xr[d];
#pragma unroll
    for (int e = 0; e < E_EXP; ++e) acc[e] += xv * gw[e * D_DIM + d];
  }
#pragma unroll
  for (int off = 16; off > 0; off >>= 1) {
#pragma unroll
    for (int e = 0; e < E_EXP; ++e) acc[e] += __shfl_xor(acc[e], off, 32);
  }
  // fp32 softmax over 8 experts (all lanes redundantly)
  float m = acc[0];
#pragma unroll
  for (int e = 1; e < E_EXP; ++e) m = fmaxf(m, acc[e]);
  float p[E_EXP], s = 0.f;
#pragma unroll
  for (int e = 0; e < E_EXP; ++e) { p[e] = __expf(acc[e] - m); s += p[e]; }
  float inv = 1.f / s;
#pragma unroll
  for (int e = 0; e < E_EXP; ++e) p[e] *= inv;
  // top-2 (first-max on ties, matches top_k)
  int i0 = 0;
#pragma unroll
  for (int e = 1; e < E_EXP; ++e) if (p[e] > p[i0]) i0 = e;
  int i1 = (i0 == 0) ? 1 : 0;
#pragma unroll
  for (int e = 0; e < E_EXP; ++e) if (e != i0 && p[e] > p[i1]) i1 = e;
  float w0 = p[i0], w1 = p[i1], wsum = w0 + w1;
  if (lane == 0) {
#pragma unroll
    for (int e = 0; e < E_EXP; ++e) probs_out[(size_t)t * E_EXP + e] = p[e];
    tki[t * 2 + 0] = i0; tki[t * 2 + 1] = i1;
    tkw[t * 2 + 0] = w0 / wsum; tkw[t * 2 + 1] = w1 / wsum;
  }
}

// ---------------- deterministic per-expert token compaction ----------------
__global__ void compact_kernel(const int* __restrict__ tki, const float* __restrict__ tkw,
                               int* __restrict__ elist, float* __restrict__ ewt,
                               int* __restrict__ ecnt) {
  __shared__ int scan[256];
  __shared__ int base_s;
  const int e = blockIdx.x;
  const int tid = threadIdx.x;
  if (tid == 0) base_s = 0;
  __syncthreads();
  for (int t0 = 0; t0 < T_TOK; t0 += 256) {
    int t = t0 + tid;
    int i0 = tki[t * 2 + 0], i1 = tki[t * 2 + 1];
    int match = 0; float w = 0.f;
    if (i0 == e)      { match = 1; w = tkw[t * 2 + 0]; }
    else if (i1 == e) { match = 1; w = tkw[t * 2 + 1]; }
    scan[tid] = match;
    __syncthreads();
    for (int off = 1; off < 256; off <<= 1) {     // inclusive Hillis-Steele scan
      int add = (tid >= off) ? scan[tid - off] : 0;
      __syncthreads();
      scan[tid] += add;
      __syncthreads();
    }
    int pos = base_s + scan[tid] - 1;
    int total = scan[255];
    if (match) { elist[e * T_TOK + pos] = t; ewt[e * T_TOK + pos] = w; }
    __syncthreads();
    if (tid == 0) base_s += total;
    __syncthreads();
  }
  int cnt = base_s;
  int padded = (cnt + (BM - 1)) & ~(BM - 1);
  if (tid < BM && (cnt + tid) < padded) {          // pad tile with weight-0 tokens
    elist[e * T_TOK + cnt + tid] = 0;
    ewt[e * T_TOK + cnt + tid] = 0.f;
  }
  if (tid == 0) ecnt[e] = cnt;
}

// ---------------- fused expert GLU MLP via WMMA bf16 ----------------
// Workgroup: 1024 threads (32 waves), BM=64 tokens, full D=1024 output.
// Wave wv: M-quarter mq=wv>>3 (rows mq*16..+15), column group wc=wv&7.
// F looped in chunks of 128; stage1: h=silu(X@W1^T)*(X@W3^T); stage2: y+=h@W2^T.
__global__ __launch_bounds__(1024, 1)
void moe_mlp_kernel(const unsigned short* __restrict__ xb,
                    const unsigned short* __restrict__ w1b,
                    const unsigned short* __restrict__ w3b,
                    const unsigned short* __restrict__ w2b,
                    const int* __restrict__ elist, const float* __restrict__ ewt,
                    const int* __restrict__ ecnt,
                    float* __restrict__ out) {
  const int e = blockIdx.y;
  const int mt = blockIdx.x;
  const int cnt = ecnt[e];
  const int tiles = (cnt + (BM - 1)) / BM;
  if (mt >= tiles) return;

  __shared__ __align__(16) unsigned short Xs[BM][1032];  // 64 x 1024 bf16 (+pad) ~132 KB
  __shared__ __align__(16) unsigned short Hs[BM][136];   // 64 x 128  bf16 (+pad) ~17 KB
  __shared__ int   tok_s[BM];
  __shared__ float twt_s[BM];

  const int tid  = threadIdx.x;
  const int lane = tid & 31;
  const int wv   = tid >> 5;          // 0..31
  const int mq   = wv >> 3;           // M quarter 0..3 -> rows mq*16..mq*16+15
  const int wc   = wv & 7;            // column group 0..7

  if (tid < BM) {
    tok_s[tid] = elist[e * T_TOK + mt * BM + tid];
    twt_s[tid] = ewt[e * T_TOK + mt * BM + tid];
  }
  __syncthreads();
  // gather BM token rows (bf16) into LDS, 16B chunks
  for (int i = tid; i < BM * 128; i += 1024) {
    int row = i >> 7, c = i & 127;
    *(uint4*)&Xs[row][c * 8] = *(const uint4*)(xb + (size_t)tok_s[row] * D_DIM + c * 8);
  }

  // fragment lane geometry (16-bit A 16x32 / B 32x16 / f32 C 16x16; C per ISA, A/B use k=16h+s)
  const int col = lane & 15;            // A: M row; B: N col; C: N col
  const int kb  = (lane >> 4) * 16;     // B: K base (16 contiguous K per lane)
  const int akb = (lane >> 4) * 16;     // A: K base (16 contiguous K per lane)
  const int crb = (lane >> 4) * 8;      // C: M row base
  const int arow = mq * 16 + col;       // A row in Xs/Hs for this wave

  const v8f zero8 = {0.f, 0.f, 0.f, 0.f, 0.f, 0.f, 0.f, 0.f};
  v8f yacc[8];
#pragma unroll
  for (int j = 0; j < 8; ++j) yacc[j] = zero8;

  const unsigned short* w1e = w1b + (size_t)e * F_DIM * D_DIM;
  const unsigned short* w3e = w3b + (size_t)e * F_DIM * D_DIM;
  const unsigned short* w2e = w2b + (size_t)e * D_DIM * F_DIM;

  for (int fc = 0; fc < F_DIM; fc += 128) {
    __syncthreads();  // Xs ready (iter 0) / previous Hs readers done
    // ---- stage 1: this wave owns F columns [fc+wc*16, fc+wc*16+16) ----
    const int fcol = fc + wc * 16 + col;
    v8f c1 = zero8, c3 = zero8;
    const unsigned short* b1p = w1e + (size_t)fcol * D_DIM + kb;
    const unsigned short* b3p = w3e + (size_t)fcol * D_DIM + kb;
    if (fc + 128 < F_DIM) {  // prefetch next chunk's W1/W3 rows (global_prefetch_b8)
      __builtin_prefetch(b1p + (size_t)128 * D_DIM, 0, 1);
      __builtin_prefetch(b3p + (size_t)128 * D_DIM, 0, 1);
    }
#pragma unroll 4
    for (int k0 = 0; k0 < D_DIM; k0 += 32) {
      FragB16 a, b1, b3;
      a.q[0] = *(const uint4*)&Xs[arow][k0 + akb];
      a.q[1] = *(const uint4*)&Xs[arow][k0 + akb + 8];
      b1.q[0] = *(const uint4*)(b1p + k0);
      b1.q[1] = *(const uint4*)(b1p + k0 + 8);
      b3.q[0] = *(const uint4*)(b3p + k0);
      b3.q[1] = *(const uint4*)(b3p + k0 + 8);
      c1 = __builtin_amdgcn_wmma_f32_16x16x32_bf16(false, a.v, false, b1.v,
                                                   (short)0, c1, false, false);
      c3 = __builtin_amdgcn_wmma_f32_16x16x32_bf16(false, a.v, false, b3.v,
                                                   (short)0, c3, false, false);
    }
    // silu(c1) * c3 -> Hs (bf16); sigmoid via v_rcp_f32 (no IEEE div chains)
#pragma unroll
    for (int r = 0; r < 8; ++r) {
      float g = c1[r];
      float sg = __builtin_amdgcn_rcpf(1.f + __expf(-g));
      float hv = g * sg * c3[r];
      Hs[mq * 16 + crb + r][wc * 16 + col] = f32_to_bf16(hv);
    }
    __syncthreads();
    // ---- stage 2: yacc[j] += Hs(16x128, quarter mq) @ W2[:, fc:fc+128]^T ----
#pragma unroll
    for (int kk = 0; kk < 128; kk += 32) {
      FragB16 a2;
      a2.q[0] = *(const uint4*)&Hs[arow][kk + akb];
      a2.q[1] = *(const uint4*)&Hs[arow][kk + akb + 8];
#pragma unroll
      for (int j = 0; j < 8; ++j) {
        const int dcol = wc * 128 + j * 16 + col;
        const unsigned short* bp = w2e + (size_t)dcol * F_DIM + fc + kk + kb;
        if (kk == 0 && fc + 128 < F_DIM)   // prefetch next chunk's W2 slice
          __builtin_prefetch(bp + 128, 0, 1);
        FragB16 b2;
        b2.q[0] = *(const uint4*)bp;
        b2.q[1] = *(const uint4*)(bp + 8);
        yacc[j] = __builtin_amdgcn_wmma_f32_16x16x32_bf16(false, a2.v, false, b2.v,
                                                          (short)0, yacc[j], false, false);
      }
    }
  }
  // scatter: out[token, d] += route_weight * y  (each token gets exactly 2 adds)
#pragma unroll
  for (int j = 0; j < 8; ++j) {
    const int d = wc * 128 + j * 16 + col;
#pragma unroll
    for (int r = 0; r < 8; ++r) {
      const int m = mq * 16 + crb + r;
      atomicAdd(&out[(size_t)tok_s[m] * D_DIM + d], twt_s[m] * yacc[j][r]);
    }
  }
}

extern "C" void kernel_launch(void* const* d_in, const int* in_sizes, int n_in,
                              void* d_out, int out_size, void* d_ws, size_t ws_size,
                              hipStream_t stream) {
  (void)in_sizes; (void)n_in; (void)out_size; (void)ws_size;
  const float* x  = (const float*)d_in[0];   // [T, D]
  const float* gw = (const float*)d_in[1];   // [E, D]
  const float* w1 = (const float*)d_in[2];   // [E, F, D]
  const float* w3 = (const float*)d_in[3];   // [E, F, D]
  const float* w2 = (const float*)d_in[4];   // [E, D, F]
  float* out   = (float*)d_out;                         // [T, D]
  float* probs = out + (size_t)T_TOK * D_DIM;           // [T, E]

  // workspace layout
  char* wsp = (char*)d_ws;
  size_t off = 0;
  auto take = [&](size_t bytes) -> char* {
    char* p = wsp + off;
    off += (bytes + 255) & ~(size_t)255;
    return p;
  };
  unsigned short* xb  = (unsigned short*)take((size_t)T_TOK * D_DIM * 2);
  unsigned short* w1b = (unsigned short*)take((size_t)E_EXP * F_DIM * D_DIM * 2);
  unsigned short* w3b = (unsigned short*)take((size_t)E_EXP * F_DIM * D_DIM * 2);
  unsigned short* w2b = (unsigned short*)take((size_t)E_EXP * D_DIM * F_DIM * 2);
  int*   tki   = (int*)take((size_t)T_TOK * 2 * sizeof(int));
  float* tkw   = (float*)take((size_t)T_TOK * 2 * sizeof(float));
  int*   elist = (int*)take((size_t)E_EXP * T_TOK * sizeof(int));
  float* ewt   = (float*)take((size_t)E_EXP * T_TOK * sizeof(float));
  int*   ecnt  = (int*)take(E_EXP * sizeof(int));

  // 1) zero MoE output (atomic accumulation target)
  zero_kernel<<<4096, 256, 0, stream>>>((float4*)out, (long long)T_TOK * D_DIM / 4);

  // 2) fp32 -> bf16 conversions
  cvt_bf16_kernel<<<4096, 256, 0, stream>>>((const float2*)x,  (unsigned*)xb,
                                            (long long)T_TOK * D_DIM / 2);
  cvt_bf16_kernel<<<8192, 256, 0, stream>>>((const float2*)w1, (unsigned*)w1b,
                                            (long long)E_EXP * F_DIM * D_DIM / 2);
  cvt_bf16_kernel<<<8192, 256, 0, stream>>>((const float2*)w3, (unsigned*)w3b,
                                            (long long)E_EXP * F_DIM * D_DIM / 2);
  cvt_bf16_kernel<<<8192, 256, 0, stream>>>((const float2*)w2, (unsigned*)w2b,
                                            (long long)E_EXP * D_DIM * F_DIM / 2);

  // 3) router (softmax output + top-2 routing), one wave per token
  router_kernel<<<T_TOK / 8, 256, 0, stream>>>(x, gw, probs, tki, tkw);

  // 4) deterministic per-expert compaction (pads lists to BM multiple)
  compact_kernel<<<E_EXP, 256, 0, stream>>>(tki, tkw, elist, ewt, ecnt);

  // 5) fused expert GLU MLP + weighted scatter
  dim3 grid(T_TOK / BM, E_EXP);
  moe_mlp_kernel<<<grid, 1024, 0, stream>>>(xb, w1b, w3b, w2b, elist, ewt, ecnt, out);
}
